// CNNLSTM_24137716204033
// MI455X (gfx1250) — matrix operand order repaired and checked by reading, not compiled, not run on hardware
//
#include <hip/hip_runtime.h>

typedef __attribute__((ext_vector_type(16))) _Float16 v16h;
typedef __attribute__((ext_vector_type(8)))  _Float16 half8;
typedef __attribute__((ext_vector_type(4)))  _Float16 half4;
typedef __attribute__((ext_vector_type(8)))  float    v8f;
typedef __attribute__((ext_vector_type(4)))  float    f4;

#define ACT_NONE 0
#define ACT_RELU 1
#define ACT_SIG  2

__device__ __forceinline__ float sigmoidf_(float x) { return 1.0f / (1.0f + __expf(-x)); }

// ---------------------------------------------------------------------------
// Generic WMMA GEMM:  C[M,Npad] = act( A[M,K] * B[K,Npad] + bias )
//   MODE 0: A = plain f32 matrix, row stride lda (M-guarded, clamped loads)
//   MODE 1: A = im2col of NHWC f32 tensor, stride-2 conv k4 p1
//   MODE 2: A = im2col of NHWC f32 tensor, lhs-dilated (transposed) conv
//   ACC  1: accumulate into existing C
// Block: 256 threads = 8 waves; macro-tile 64(M) x 32(N); K step 64 with two
// v_wmma_f32_16x16x32_f16 per barrier interval.  K % 64 == 0 required.
// ---------------------------------------------------------------------------
template <int MODE, int ACT, int ACC>
__global__ __launch_bounds__(256)
void wmma_gemm_k(const float* __restrict__ A, long lda,
                 const _Float16* __restrict__ Bw,
                 const float* __restrict__ bias, int nbias,
                 float* __restrict__ C, long ldc,
                 int M, int Npad, int K,
                 int IH, int IW, int Cin, int owS, int ohS) {
  __shared__ _Float16 As[64 * 64];   // [m][k]   8 KB
  __shared__ _Float16 Bst[32 * 64];  // [n][k]   4 KB (transposed)

  const int t    = threadIdx.x;
  const int lane = t & 31;
  const int wid  = t >> 5;          // 0..7
  const int wm   = wid & 3;         // M sub-tile
  const int wn   = wid >> 2;        // N sub-tile
  const int lo   = lane & 15;
  const int hi   = (lane >> 4) & 1;
  const int tileMb = blockIdx.x * 64;
  const int tileNb = blockIdx.y * 32;

  const int mb   = hi ? 8 : 0;
  const int row0 = tileMb + wm * 16 + mb;      // + v
  const int col  = tileNb + wn * 16 + lo;

  v8f c;
  if (ACC) {
#pragma unroll
    for (int v = 0; v < 8; ++v) {
      int r  = row0 + v;
      int rc = r < M ? r : (M - 1);
      int cc = col < Npad ? col : (Npad - 1);
      float cv = C[(long)rc * ldc + cc];
      c[v] = (r < M && col < Npad) ? cv : 0.0f;
    }
  } else {
#pragma unroll
    for (int v = 0; v < 8; ++v) c[v] = 0.0f;
  }

  const int owMask = (1 << owS) - 1;
  const int ohMask = (1 << ohS) - 1;

  for (int k0 = 0; k0 < K; k0 += 64) {
    __syncthreads();

    // ---- stage A tile: 64 x 64 f16 (16 groups-of-4 per row, 4 per thread) ----
#pragma unroll
    for (int i = 0; i < 4; ++i) {
      int g  = t + i * 256;
      int r  = g >> 4;               // 0..63
      int kg = (g & 15) * 4;         // 0,4,..,60
      int grow = tileMb + r;
      int gk   = k0 + kg;
      half4 hv;
      if (MODE == 0) {
        int ok = grow < M;
        int gr = ok ? grow : 0;
        f4 v = *(const f4*)&A[(long)gr * lda + gk];
        float m = ok ? 1.0f : 0.0f;
#pragma unroll
        for (int j = 0; j < 4; ++j) hv[j] = (_Float16)(v[j] * m);
      } else {
        // decompose output position (shared by the 4 k's; kg%4==0 keeps ci,ky fixed)
        int ox = grow & owMask;
        int tt = grow >> owS;
        int oy = tt & ohMask;
        int n  = tt >> ohS;
        int ci = gk >> 4;
        int ky = (gk >> 2) & 3;
        int okC = ci < Cin;
        int cic = okC ? ci : (Cin - 1);
        int iy, okY;
        if (MODE == 1) {
          iy  = 2 * oy - 1 + ky;
          okY = ((unsigned)iy < (unsigned)IH);
        } else {
          int sy = oy - 2 + ky;
          okY = (sy >= 0) & !(sy & 1);
          iy  = sy >> 1;
          okY &= (iy < IH);
        }
        int iyc = okY ? iy : 0;
        const float* base = A + (((long)n * IH + iyc) * IW) * Cin + cic;
#pragma unroll
        for (int j = 0; j < 4; ++j) {
          int ix, okX;
          if (MODE == 1) {
            ix  = 2 * ox - 1 + j;
            okX = ((unsigned)ix < (unsigned)IW);
          } else {
            int sx = ox - 2 + j;
            okX = (sx >= 0) & !(sx & 1);
            ix  = sx >> 1;
            okX &= (ix < IW);
          }
          int ixc = okX ? ix : 0;
          float v = base[(long)ixc * Cin];
          hv[j] = (_Float16)((okY & okX & okC) ? v : 0.0f);
        }
      }
      *(half4*)&As[r * 64 + kg] = hv;
    }

    // ---- stage B tile (transposed in LDS): Bst[n][k], 64 k-rows x 32 cols ----
#pragma unroll
    for (int i = 0; i < 2; ++i) {
      int n  = t & 31;                     // 0..31
      int kg = (t >> 5) * 4 + i * 32;      // 0,4,..,28 / 32,..,60
      int gc  = tileNb + n;
      int gcc = gc < Npad ? gc : (Npad - 1);
      half4 hv;
#pragma unroll
      for (int j = 0; j < 4; ++j) {
        _Float16 bv = Bw[(long)(k0 + kg + j) * Npad + gcc];
        hv[j] = (gc < Npad) ? bv : (_Float16)0.0f;
      }
      *(half4*)&Bst[n * 64 + kg] = hv;
    }

    if (k0 + 64 < K && t < 32)  // hint next B tile (global_prefetch_b8)
      __builtin_prefetch(&Bw[(long)(k0 + 64) * Npad + tileNb + lane], 0, 1);
    __syncthreads();

    // ---- fragments (vector LDS loads) + two WMMAs per interval ----
    const half8* Arow = (const half8*)&As[(wm * 16 + lo) * 64];
    const half8* Brow = (const half8*)&Bst[(wn * 16 + lo) * 64];
    v16h af, bf;
    // K = k0 .. k0+31
    ((half8*)&af)[0] = Arow[hi];
    ((half8*)&af)[1] = Arow[hi + 2];
    ((half8*)&bf)[0] = Brow[hi];
    ((half8*)&bf)[1] = Brow[hi + 2];
    c = __builtin_amdgcn_wmma_f32_16x16x32_f16(false, af, false, bf, (short)0, c,
                                               false, false);
    // K = k0+32 .. k0+63
    ((half8*)&af)[0] = Arow[hi + 4];
    ((half8*)&af)[1] = Arow[hi + 6];
    ((half8*)&bf)[0] = Brow[hi + 4];
    ((half8*)&bf)[1] = Brow[hi + 6];
    c = __builtin_amdgcn_wmma_f32_16x16x32_f16(false, af, false, bf, (short)0, c,
                                               false, false);
  }

  const float bv = (bias != nullptr && col < nbias) ? bias[col] : 0.0f;
#pragma unroll
  for (int v = 0; v < 8; ++v) {
    int r = row0 + v;
    float x = c[v] + bv;
    if (ACT == ACT_RELU)     x = x > 0.0f ? x : 0.0f;
    else if (ACT == ACT_SIG) x = sigmoidf_(x);
    if (r < M && col < Npad) C[(long)r * ldc + col] = x;
  }
}

// ---------- weight packing: f32 -> f16 B-matrices [K, Npad] ----------
__global__ void pack_conv_k(const float* __restrict__ W, _Float16* __restrict__ Bm,
                            int Cin, int Cout, int Npad, int Kpad) {
  int idx = blockIdx.x * 256 + threadIdx.x;
  int total = Kpad * Npad;
  if (idx >= total) return;
  int k = idx / Npad, n = idx % Npad;
  float v = 0.0f;
  if (n < Cout && k < Cin * 16) {
    int ci = k >> 4, kk = k & 15;
    v = W[((long)n * Cin + ci) * 16 + kk];
  }
  Bm[idx] = (_Float16)v;
}

__global__ void pack_fc_k(const float* __restrict__ W, _Float16* __restrict__ Bm,
                          int In, int Out, int Npad) {
  int idx = blockIdx.x * 256 + threadIdx.x;
  int total = In * Npad;
  if (idx >= total) return;
  int k = idx / Npad, n = idx % Npad;
  Bm[idx] = (_Float16)((n < Out) ? W[(long)n * In + k] : 0.0f);
}

__global__ void pack_fc_permin_k(const float* __restrict__ W, _Float16* __restrict__ Bm,
                                 int In, int Out, int Npad, int Cch, int HW) {
  int idx = blockIdx.x * 256 + threadIdx.x;
  int total = In * Npad;
  if (idx >= total) return;
  int k = idx / Npad, n = idx % Npad;
  float v = 0.0f;
  if (n < Out) {
    int hw = k / Cch, c = k % Cch;
    v = W[(long)n * In + c * HW + hw];
  }
  Bm[idx] = (_Float16)v;
}

__global__ void pack_fc_permout_k(const float* __restrict__ W, _Float16* __restrict__ Bm,
                                  int In, int Out, int Npad, int Cch, int HW) {
  int idx = blockIdx.x * 256 + threadIdx.x;
  int total = In * Npad;
  if (idx >= total) return;
  int k = idx / Npad, n = idx % Npad;
  float v = 0.0f;
  if (n < Out) {
    int hw = n / Cch, c = n % Cch;
    v = W[((long)c * HW + hw) * In + k];
  }
  Bm[idx] = (_Float16)v;
}

__global__ void perm_bias_k(const float* __restrict__ b, float* __restrict__ o,
                            int Cch, int HW) {
  int idx = blockIdx.x * 256 + threadIdx.x;
  if (idx >= Cch * HW) return;
  int hw = idx / Cch, c = idx % Cch;
  o[idx] = b[c * HW + hw];
}

__global__ void vec_add_k(const float* __restrict__ a, const float* __restrict__ b,
                          float* __restrict__ o, int n) {
  int idx = blockIdx.x * 256 + threadIdx.x;
  if (idx < n) o[idx] = a[idx] + b[idx];
}

// video [N,3,64,64] f32 NCHW -> [N,64,64,3] NHWC
__global__ void video_nhwc_k(const float* __restrict__ v, float* __restrict__ o, int total) {
  int idx = blockIdx.x * 256 + threadIdx.x;
  if (idx >= total) return;
  int ch = idx % 3;
  int x  = (idx / 3) & 63;
  int y  = (idx / 192) & 63;
  int n  = idx / 12288;
  o[idx] = v[(((long)n * 3 + ch) * 64 + y) * 64 + x];
}

// [N,64,64,16] NHWC (3 valid ch) -> d_out [N,3,64,64]
__global__ void nhwc_out_k(const float* __restrict__ X, float* __restrict__ o, int total) {
  int idx = blockIdx.x * 256 + threadIdx.x;
  if (idx >= total) return;
  int x = idx & 63;
  int y = (idx >> 6) & 63;
  int c = (idx >> 12) % 3;
  int n = idx / 12288;
  o[idx] = X[(((long)n * 64 + y) * 64 + x) * 16 + c];
}

// LSTM pointwise: G[16,2048] (i,f,g,o), h/c [16,512]
__global__ void lstm_point_k(const float* __restrict__ G, float* __restrict__ h,
                             float* __restrict__ c) {
  int idx = blockIdx.x * 256 + threadIdx.x;
  if (idx >= 16 * 512) return;
  int b = idx >> 9, j = idx & 511;
  const float* g = G + (long)b * 2048;
  float ig = sigmoidf_(g[j]);
  float fg = sigmoidf_(g[512 + j]);
  float gg = tanhf(g[1024 + j]);
  float og = sigmoidf_(g[1536 + j]);
  float cn = fg * c[idx] + ig * gg;
  c[idx] = cn;
  h[idx] = og * tanhf(cn);
}

extern "C" void kernel_launch(void* const* d_in, const int* in_sizes, int n_in,
                              void* d_out, int out_size, void* d_ws, size_t ws_size,
                              hipStream_t stream) {
  (void)in_sizes; (void)n_in; (void)out_size; (void)ws_size;
  const float* video  = (const float*)d_in[0];
  const float* ec1_w  = (const float*)d_in[2];  const float* ec1_b = (const float*)d_in[3];
  const float* ec2_w  = (const float*)d_in[4];  const float* ec2_b = (const float*)d_in[5];
  const float* ec3_w  = (const float*)d_in[6];  const float* ec3_b = (const float*)d_in[7];
  const float* ec4_w  = (const float*)d_in[8];  const float* ec4_b = (const float*)d_in[9];
  const float* fcmu_w = (const float*)d_in[10]; const float* fcmu_b = (const float*)d_in[11];
  const float* dfc_w  = (const float*)d_in[12]; const float* dfc_b  = (const float*)d_in[13];
  const float* dt1_w  = (const float*)d_in[14]; const float* dt1_b  = (const float*)d_in[15];
  const float* dt2_w  = (const float*)d_in[16]; const float* dt2_b  = (const float*)d_in[17];
  const float* dt3_w  = (const float*)d_in[18]; const float* dt3_b  = (const float*)d_in[19];
  const float* dt4_w  = (const float*)d_in[20]; const float* dt4_b  = (const float*)d_in[21];
  const float* wih_e  = (const float*)d_in[22]; const float* whh_e  = (const float*)d_in[23];
  const float* bih_e  = (const float*)d_in[24]; const float* bhh_e  = (const float*)d_in[25];
  const float* wih_d  = (const float*)d_in[26]; const float* whh_d  = (const float*)d_in[27];
  const float* bih_d  = (const float*)d_in[28]; const float* bhh_d  = (const float*)d_in[29];
  const float* fc_w   = (const float*)d_in[30]; const float* fc_b   = (const float*)d_in[31];
  float* out = (float*)d_out;

  char* ws = (char*)d_ws;
  size_t off = 0;
  auto alloc = [&](size_t bytes) { size_t o = off; off += (bytes + 255) & ~(size_t)255; return o; };

  float* P0   = (float*)(ws + alloc(256UL * 64 * 64 * 16 * 4));  // 64 MB ping
  float* P1   = (float*)(ws + alloc(256UL * 32 * 32 * 32 * 4));  // 32 MB pong
  float* vnh  = (float*)(ws + alloc(256UL * 64 * 64 * 3 * 4));
  float* z    = (float*)(ws + alloc(256UL * 128 * 4));
  float* zd   = (float*)(ws + alloc(256UL * 128 * 4));
  float* h    = (float*)(ws + alloc(16UL * 512 * 4));
  float* cst  = (float*)(ws + alloc(16UL * 512 * 4));
  float* G    = (float*)(ws + alloc(16UL * 2048 * 4));
  float* bse  = (float*)(ws + alloc(2048UL * 4));
  float* bsd  = (float*)(ws + alloc(2048UL * 4));
  float* dbias= (float*)(ws + alloc(4096UL * 4));
  _Float16* ec1p  = (_Float16*)(ws + alloc(64UL * 32 * 2));
  _Float16* ec2p  = (_Float16*)(ws + alloc(512UL * 64 * 2));
  _Float16* ec3p  = (_Float16*)(ws + alloc(1024UL * 128 * 2));
  _Float16* ec4p  = (_Float16*)(ws + alloc(2048UL * 256 * 2));
  _Float16* fcmup = (_Float16*)(ws + alloc(4096UL * 128 * 2));
  _Float16* dfcp  = (_Float16*)(ws + alloc(128UL * 4096 * 2));
  _Float16* dt1p  = (_Float16*)(ws + alloc(4096UL * 128 * 2));
  _Float16* dt2p  = (_Float16*)(ws + alloc(2048UL * 64 * 2));
  _Float16* dt3p  = (_Float16*)(ws + alloc(1024UL * 32 * 2));
  _Float16* dt4p  = (_Float16*)(ws + alloc(512UL * 16 * 2));
  _Float16* wihep = (_Float16*)(ws + alloc(128UL * 2048 * 2));
  _Float16* whhep = (_Float16*)(ws + alloc(512UL * 2048 * 2));
  _Float16* wihdp = (_Float16*)(ws + alloc(128UL * 2048 * 2));
  _Float16* whhdp = (_Float16*)(ws + alloc(512UL * 2048 * 2));
  _Float16* fcwp  = (_Float16*)(ws + alloc(512UL * 128 * 2));

  auto cdiv = [](long a, long b) { return (unsigned)((a + b - 1) / b); };
  auto gemm = [&](int mode, int act, int acc,
                  const float* A, long lda, const _Float16* Bw, const float* bias, int nbias,
                  float* C, long ldc, int M, int Npad, int K,
                  int IH, int IW, int Cin, int owS, int ohS) {
    dim3 grid((unsigned)((M + 63) / 64), (unsigned)((Npad + 31) / 32), 1);
    dim3 blk(256, 1, 1);
#define LK(MO, AC, CC)                                                                 \
  wmma_gemm_k<MO, AC, CC><<<grid, blk, 0, stream>>>(A, lda, Bw, bias, nbias, C, ldc,   \
                                                    M, Npad, K, IH, IW, Cin, owS, ohS)
    if (mode == 0) {
      if (acc)                      LK(0, ACT_NONE, 1);
      else if (act == ACT_RELU)     LK(0, ACT_RELU, 0);
      else                          LK(0, ACT_NONE, 0);
    } else if (mode == 1) {
      LK(1, ACT_RELU, 0);
    } else {
      if (act == ACT_SIG)           LK(2, ACT_SIG, 0);
      else                          LK(2, ACT_RELU, 0);
    }
#undef LK
  };

  hipMemsetAsync(h, 0, 16 * 512 * 4, stream);
  hipMemsetAsync(cst, 0, 16 * 512 * 4, stream);

  // ---- pack all weights to f16 [K, Npad] ----
  pack_conv_k<<<cdiv(64L * 32, 256), 256, 0, stream>>>(ec1_w, ec1p, 3, 32, 32, 64);
  pack_conv_k<<<cdiv(512L * 64, 256), 256, 0, stream>>>(ec2_w, ec2p, 32, 64, 64, 512);
  pack_conv_k<<<cdiv(1024L * 128, 256), 256, 0, stream>>>(ec3_w, ec3p, 64, 128, 128, 1024);
  pack_conv_k<<<cdiv(2048L * 256, 256), 256, 0, stream>>>(ec4_w, ec4p, 128, 256, 256, 2048);
  pack_conv_k<<<cdiv(4096L * 128, 256), 256, 0, stream>>>(dt1_w, dt1p, 256, 128, 128, 4096);
  pack_conv_k<<<cdiv(2048L * 64, 256), 256, 0, stream>>>(dt2_w, dt2p, 128, 64, 64, 2048);
  pack_conv_k<<<cdiv(1024L * 32, 256), 256, 0, stream>>>(dt3_w, dt3p, 64, 32, 32, 1024);
  pack_conv_k<<<cdiv(512L * 16, 256), 256, 0, stream>>>(dt4_w, dt4p, 32, 3, 16, 512);
  pack_fc_permin_k<<<cdiv(4096L * 128, 256), 256, 0, stream>>>(fcmu_w, fcmup, 4096, 128, 128, 256, 16);
  pack_fc_permout_k<<<cdiv(128L * 4096, 256), 256, 0, stream>>>(dfc_w, dfcp, 128, 4096, 4096, 256, 16);
  perm_bias_k<<<cdiv(4096, 256), 256, 0, stream>>>(dfc_b, dbias, 256, 16);
  pack_fc_k<<<cdiv(128L * 2048, 256), 256, 0, stream>>>(wih_e, wihep, 128, 2048, 2048);
  pack_fc_k<<<cdiv(512L * 2048, 256), 256, 0, stream>>>(whh_e, whhep, 512, 2048, 2048);
  pack_fc_k<<<cdiv(128L * 2048, 256), 256, 0, stream>>>(wih_d, wihdp, 128, 2048, 2048);
  pack_fc_k<<<cdiv(512L * 2048, 256), 256, 0, stream>>>(whh_d, whhdp, 512, 2048, 2048);
  pack_fc_k<<<cdiv(512L * 128, 256), 256, 0, stream>>>(fc_w, fcwp, 512, 128, 128);
  vec_add_k<<<cdiv(2048, 256), 256, 0, stream>>>(bih_e, bhh_e, bse, 2048);
  vec_add_k<<<cdiv(2048, 256), 256, 0, stream>>>(bih_d, bhh_d, bsd, 2048);

  // ---- encoder CNN (implicit-GEMM WMMA convs, NHWC f32 activations) ----
  video_nhwc_k<<<cdiv(256L * 64 * 64 * 3, 256), 256, 0, stream>>>(video, vnh, 256 * 64 * 64 * 3);
  gemm(1, ACT_RELU, 0, vnh, 0, ec1p, ec1_b, 32, P0, 32, 262144, 32, 64, 64, 64, 3, 5, 5);
  gemm(1, ACT_RELU, 0, P0, 0, ec2p, ec2_b, 64, P1, 64, 65536, 64, 512, 32, 32, 32, 4, 4);
  gemm(1, ACT_RELU, 0, P1, 0, ec3p, ec3_b, 128, P0, 128, 16384, 128, 1024, 16, 16, 64, 3, 3);
  gemm(1, ACT_RELU, 0, P0, 0, ec4p, ec4_b, 256, P1, 256, 4096, 256, 2048, 8, 8, 128, 2, 2);
  // fcmu: [256,4096] x [4096,128] (HWC->CHW folded into weight pack)
  gemm(0, ACT_NONE, 0, P1, 4096, fcmup, fcmu_b, 128, z, 128, 256, 128, 4096, 0, 0, 1, 0, 0);

  // ---- encoder LSTM over T=16 (rows of z are n = b*16 + t -> lda 2048) ----
  for (int t = 0; t < 16; ++t) {
    gemm(0, ACT_NONE, 0, z + t * 128, 2048, wihep, bse, 2048, G, 2048, 16, 2048, 128, 0, 0, 1, 0, 0);
    gemm(0, ACT_NONE, 1, h, 512, whhep, nullptr, 0, G, 2048, 16, 2048, 512, 0, 0, 1, 0, 0);
    lstm_point_k<<<32, 256, 0, stream>>>(G, h, cst);
  }

  // ---- autoregressive decoder LSTM (h,c carry over from encoder) ----
  for (int t = 0; t < 16; ++t) {
    const float* x = (t == 0) ? (z + 15 * 128) : (zd + (t - 1) * 128);
    gemm(0, ACT_NONE, 0, x, 2048, wihdp, bsd, 2048, G, 2048, 16, 2048, 128, 0, 0, 1, 0, 0);
    gemm(0, ACT_NONE, 1, h, 512, whhdp, nullptr, 0, G, 2048, 16, 2048, 512, 0, 0, 1, 0, 0);
    lstm_point_k<<<32, 256, 0, stream>>>(G, h, cst);
    gemm(0, ACT_NONE, 0, h, 512, fcwp, fc_b, 128, zd + t * 128, 2048, 16, 128, 512, 0, 0, 1, 0, 0);
  }

  // ---- decoder CNN ----
  gemm(0, ACT_RELU, 0, zd, 128, dfcp, dbias, 4096, P0, 4096, 256, 4096, 128, 0, 0, 1, 0, 0);
  gemm(2, ACT_RELU, 0, P0, 0, dt1p, dt1_b, 128, P1, 128, 16384, 128, 4096, 4, 4, 256, 3, 3);
  gemm(2, ACT_RELU, 0, P1, 0, dt2p, dt2_b, 64, P0, 64, 65536, 64, 2048, 8, 8, 128, 4, 4);
  gemm(2, ACT_RELU, 0, P0, 0, dt3p, dt3_b, 32, P1, 32, 262144, 32, 1024, 16, 16, 64, 5, 5);
  gemm(2, ACT_SIG, 0, P1, 0, dt4p, dt4_b, 3, P0, 16, 1048576, 16, 512, 32, 32, 32, 6, 6);

  nhwc_out_k<<<cdiv(256L * 3 * 64 * 64, 256), 256, 0, stream>>>(P0, out, 256 * 3 * 64 * 64);
}
